// ConvOffset2D_7584912245429
// MI455X (gfx1250) — matrix operand order repaired and checked by reading, not compiled, not run on hardware
//
#include <hip/hip_runtime.h>
#include <hip/hip_bf16.h>

// ---------------------------------------------------------------------------
// ConvOffset2D (deformable-conv offset sampling) for MI455X / gfx1250.
// Phase 1: 3x3 SAME conv as bf16 WMMA GEMM (M=65536, K=1152, N=256),
//          B panels staged per-tap via TENSOR_LOAD_TO_LDS (TDM), double-buffered.
//          Each wave: 64M x 32N tile (4x2 WMMA subtiles), K-chunk software pipeline.
// Phase 2: channel-major bilinear sampling (memory bound).
// ---------------------------------------------------------------------------

typedef unsigned short u16;
typedef unsigned int   u32;
typedef __attribute__((ext_vector_type(16))) __bf16 v16bf;
typedef __attribute__((ext_vector_type(8)))  float  v8f;
typedef __attribute__((ext_vector_type(4)))  u16    u16x4;
typedef __attribute__((ext_vector_type(4)))  u32    u32x4;
typedef __attribute__((ext_vector_type(8)))  int    i32x8;
typedef __attribute__((ext_vector_type(4)))  int    i32x4;

union Frag {
    uint4 u[2];   // two b128 LDS loads
    v16bf v;      // WMMA operand
};

#define B_   16
#define H_   64
#define W_   64
#define C_   128
#define N_   256                 // 2*C
#define XL_BYTES  50688          // 3*66*128 bf16 halo slab
#define BPANEL_ELEMS 32768       // 256*128 bf16 per tap
#define BPANEL_BYTES 65536
#define SMEM_BYTES (XL_BYTES + 2 * BPANEL_BYTES)   // 181760 B < 320 KB

#if defined(__AMDGCN__) && __has_builtin(__builtin_amdgcn_tensor_load_to_lds) && \
    __has_builtin(__builtin_amdgcn_s_wait_tensorcnt)
#define HAVE_TDM 1
#else
#define HAVE_TDM 0
#endif

static __device__ __forceinline__ u16 f2bf(float f) {
    u32 u = __float_as_uint(f);
    u += 0x7FFFu + ((u >> 16) & 1u);   // round-to-nearest-even
    return (u16)(u >> 16);
}

#if HAVE_TDM
// Issue a TDM copy of one 64 KB B panel (32768 x 2B contiguous) into LDS.
static __device__ __forceinline__ void tdm_load_b_panel(const u16* src, u32 lds_off) {
    unsigned long long ga = (unsigned long long)(const void*)src;
    u32x4 g0;
    g0[0] = 1u;                                   // count=1 (valid), user mode
    g0[1] = lds_off;                              // lds_addr [63:32]
    g0[2] = (u32)ga;                              // global_addr [95:64]
    g0[3] = (u32)((ga >> 32) & 0x01FFFFFFull)     // global_addr [120:96]
          | (2u << 30);                           // type=2 ("image")
    i32x8 g1;
    g1[0] = (int)(1u << 16);                      // data_size=1 -> 2 bytes/elem
    g1[1] = (int)((BPANEL_ELEMS & 0xFFFFu) << 16);// tensor_dim0 lo16 @ [31:16]
    g1[2] = (int)((BPANEL_ELEMS >> 16) | (1u << 16)); // dim0 hi16 | tensor_dim1=1
    g1[3] = (int)((u32)BPANEL_ELEMS << 16);       // tensor_dim1 hi=0 | tile_dim0
    g1[4] = 1;                                    // tile_dim1=1, tile_dim2=0
    g1[5] = BPANEL_ELEMS;                         // tensor_dim0_stride lo32
    g1[6] = 0;                                    // stride hi | dim1_stride lo
    g1[7] = 0;
    i32x4 z4 = {0, 0, 0, 0};
#if __clang_major__ >= 23
    i32x8 z8 = {0, 0, 0, 0, 0, 0, 0, 0};
    __builtin_amdgcn_tensor_load_to_lds(g0, g1, z4, z4, z8, 0);
#else
    __builtin_amdgcn_tensor_load_to_lds(g0, g1, z4, z4, 0);
#endif
}
#endif

// --- prep: W (3,3,128,256) f32 HWIO -> Wb[tap][n][k] bf16 ------------------
__global__ __launch_bounds__(256) void convert_W_kernel(
    const float* __restrict__ W, u16* __restrict__ Wb) {
    int idx = blockIdx.x * 256 + threadIdx.x;        // < 9*256*128 = 294912
    int t = idx >> 15;                               // tap  (kh*3+kw)
    int n = (idx >> 7) & 255;                        // out channel
    int k = idx & 127;                               // in channel
    Wb[idx] = f2bf(W[(t * C_ + k) * N_ + n]);
}

// --- phase 1: conv-as-GEMM with v_wmma_f32_16x16x32_bf16 -------------------
__global__ __launch_bounds__(256) void conv_gemm_kernel(
    const float* __restrict__ x, const u16* __restrict__ Wb,
    const float* __restrict__ bias, float* __restrict__ offs) {
    extern __shared__ char smem[];
    u16* xl = (u16*)smem;                 // [3][66][128] bf16, halo-padded row slab
    u16* Bl = (u16*)(smem + XL_BYTES);    // 2 x [256][128] bf16 B panels

    const int tid = threadIdx.x;
    const int wg  = blockIdx.x;           // 1024 = B*H
    const int bi  = wg >> 6;
    const int row = wg & 63;
    const int wave = tid >> 5;

#if HAVE_TDM
    if (wave == 0) tdm_load_b_panel(Wb, XL_BYTES);   // prefetch tap 0 -> buf 0
#endif

    // zero whole slab (covers SAME padding: cols -1/64 and out-of-image rows)
    u32* xz = (u32*)xl;
    for (int e = tid; e < (XL_BYTES / 4); e += 256) xz[e] = 0u;
    __syncthreads();

    // fill interior: convert f32 -> bf16 while staging into LDS
    for (int r = 0; r < 3; ++r) {
        int ir = row - 1 + r;
        if (ir < 0 || ir > 63) continue;
        const float* src = x + (((size_t)bi * H_ + ir) * W_) * C_;  // 8192 f32
        u16* dst = xl + (r * 66 + 1) * C_;
        for (int e = tid * 4; e < W_ * C_; e += 256 * 4) {
            float4 v = *(const float4*)(src + e);
            u16x4 o;
            o.x = f2bf(v.x); o.y = f2bf(v.y); o.z = f2bf(v.z); o.w = f2bf(v.w);
            *(u16x4*)(dst + e) = o;
        }
    }

    const int lane  = tid & 31;
    const int laneM = lane & 15;          // M (A) or N (B/C) index
    const int laneH = lane >> 4;          // K-half selector per ISA layouts
    const int n0 = wave * 32;             // each wave: 2 N-subtiles, all 4 M-subtiles

    v8f acc[4][2];
    v8f zero;
    #pragma unroll
    for (int e = 0; e < 8; ++e) zero[e] = 0.0f;
    #pragma unroll
    for (int mt = 0; mt < 4; ++mt) {
        acc[mt][0] = zero;
        acc[mt][1] = zero;
    }

    for (int t = 0; t < 9; ++t) {         // 3x3 taps, K = 9*128
        const int dr = t / 3;             // kh
        const int dj = t % 3;             // kw
#if HAVE_TDM
        if (wave == 0) __builtin_amdgcn_s_wait_tensorcnt(0);  // buf[t&1] filled
        __syncthreads();                  // panel visible; prev compute drained
        if (t < 8 && wave == 0)           // overlap: fetch tap t+1 into other buf
            tdm_load_b_panel(Wb + (size_t)(t + 1) * BPANEL_ELEMS,
                             XL_BYTES + (u32)((t + 1) & 1) * BPANEL_BYTES);
        const u16* Bt = Bl + (t & 1) * BPANEL_ELEMS;
#else
        __syncthreads();                  // previous tap's panel fully consumed
        {   // stage this tap's 256x128 bf16 B panel (64 KB)
            const uint4* src = (const uint4*)(Wb + (size_t)t * BPANEL_ELEMS);
            uint4* dst = (uint4*)Bl;
            for (int e = tid; e < (BPANEL_BYTES / 16); e += 256) dst[e] = src[e];
        }
        __syncthreads();
        const u16* Bt = Bl;
#endif

        // A base: halo col jj = (spatial col j) + dj, j = mt*16 + laneM
        const u16* abase = xl + (dr * 66 + laneM + dj) * C_ + laneH * 8;
        const u16* bbase = Bt + (n0 + laneM) * C_ + laneH * 8;

        // fragment loader for one 32-wide K chunk (ISA 7.12.2 16-bit layouts)
        auto load_frags = [&](Frag (&aF)[4], Frag (&bF)[2], int kc) {
            #pragma unroll
            for (int mt = 0; mt < 4; ++mt) {
                const u16* ap = abase + (mt * 16) * C_ + kc * 32;
                aF[mt].u[0] = *(const uint4*)(ap);        // K kb..kb+7
                aF[mt].u[1] = *(const uint4*)(ap + 16);   // K kb+16..kb+23
            }
            #pragma unroll
            for (int nt = 0; nt < 2; ++nt) {
                const u16* bp = bbase + (nt * 16) * C_ + kc * 32;
                bF[nt].u[0] = *(const uint4*)(bp);
                bF[nt].u[1] = *(const uint4*)(bp + 16);
            }
        };

        Frag aF[2][4];
        Frag bF[2][2];
        load_frags(aF[0], bF[0], 0);      // prologue: kc=0 fragments
        #pragma unroll
        for (int kc = 0; kc < 4; ++kc) {  // 128 channels = 4 x K32, 2-deep pipe
            const int cur = kc & 1;
            const int nxt = cur ^ 1;
            if (kc < 3) load_frags(aF[nxt], bF[nxt], kc + 1);
            #pragma unroll
            for (int nt = 0; nt < 2; ++nt) {
                #pragma unroll
                for (int mt = 0; mt < 4; ++mt) {
                    acc[mt][nt] = __builtin_amdgcn_wmma_f32_16x16x32_bf16(
                        false, aF[cur][mt].v, false, bF[cur][nt].v,
                        (short)0, acc[mt][nt], false, false);
                }
            }
        }
    }

    // store C + bias; 16x16 f32 C layout: VGPR r -> M = r + 8*laneH, N = laneM
    float* orow = offs + (((size_t)bi * H_ + row) * W_) * N_;
    #pragma unroll
    for (int nt = 0; nt < 2; ++nt) {
        int ch = n0 + nt * 16 + laneM;
        float bv = bias[ch];
        #pragma unroll
        for (int mt = 0; mt < 4; ++mt) {
            #pragma unroll
            for (int r = 0; r < 8; ++r) {
                int j = mt * 16 + r + laneH * 8;
                orow[(size_t)j * N_ + ch] = acc[mt][nt][r] + bv;
            }
        }
    }
}

// --- phase 2: channel-major bilinear sampling ------------------------------
__global__ __launch_bounds__(256) void sample_kernel(
    const float* __restrict__ x, const float* __restrict__ offs,
    float* __restrict__ out) {
    int tid = blockIdx.x * 256 + threadIdx.x;   // B*H*W*C = 8388608
    int m  = tid & 127;                          // channel
    int r  = tid >> 7;
    int jp = r & 63;  r >>= 6;
    int ip = r & 63;
    int bi = r >> 6;

    // reconstruct TF-style reshape: off[n=bi*C+m, p=ip*W+jp, k] lives at
    // batch-linear T = (m*H + ip)*2*W + 2*jp + k in the (2C,H,W) block
    float o[2];
    #pragma unroll
    for (int k = 0; k < 2; ++k) {
        int T   = (m * H_ + ip) * (2 * W_) + 2 * jp + k;
        int ch  = T >> 12;          // / (H*W)
        int rem = T & 4095;
        int ii  = rem >> 6;
        int jj  = rem & 63;
        o[k] = offs[(((size_t)bi * H_ + ii) * W_ + jj) * N_ + ch];
    }

    float c0 = o[0] + (float)ip;
    float c1 = o[1] + (float)jp;
    c0 = fminf(fmaxf(c0, 0.0f), (float)(W_ - 1));
    c1 = fminf(fmaxf(c1, 0.0f), (float)(H_ - 1));
    float fl0 = floorf(c0), fl1 = floorf(c1);
    int lt0 = (int)fl0, lt1 = (int)fl1;
    int rb0 = (int)ceilf(c0), rb1 = (int)ceilf(c1);

    const float* xb = x + ((size_t)bi * H_ * W_) * C_ + m;
    float vlt = xb[(size_t)(lt0 * W_ + lt1) * C_];
    float vrb = xb[(size_t)(rb0 * W_ + rb1) * C_];
    float vlb = xb[(size_t)(lt0 * W_ + rb1) * C_];
    float vrt = xb[(size_t)(rb0 * W_ + lt1) * C_];

    float f0 = c0 - fl0;
    float f1 = c1 - fl1;
    float vt = vlt + (vrt - vlt) * f0;
    float vb = vlb + (vrb - vlb) * f0;
    out[tid] = vt + (vb - vt) * f1;
}

// ---------------------------------------------------------------------------
extern "C" void kernel_launch(void* const* d_in, const int* in_sizes, int n_in,
                              void* d_out, int out_size, void* d_ws, size_t ws_size,
                              hipStream_t stream) {
    const float* x = (const float*)d_in[0];   // (16,64,64,128) f32
    const float* W = (const float*)d_in[1];   // (3,3,128,256)  f32
    const float* b = (const float*)d_in[2];   // (256,)         f32
    float* out = (float*)d_out;               // (16,64,64,128) f32

    // workspace layout: offsets f32 (64 MB) | Wb bf16 (576 KB)
    float* offs = (float*)d_ws;
    u16*   Wb   = (u16*)((char*)d_ws + (size_t)B_ * H_ * W_ * N_ * sizeof(float));

    convert_W_kernel<<<(9 * C_ * N_) / 256, 256, 0, stream>>>(W, Wb);
    conv_gemm_kernel<<<B_ * H_, 256, SMEM_BYTES, stream>>>(x, Wb, b, offs);
    sample_kernel<<<(B_ * H_ * W_ * C_) / 256, 256, 0, stream>>>(x, offs, out);
}
